// SparseBayesianLinear_37821482009033
// MI455X (gfx1250) — matrix operand order repaired and checked by reading, not compiled, not run on hardware
//
#include <hip/hip_runtime.h>
#include <cstdint>

// ---------------------------------------------------------------------------
// SparseBayesianLinear fused kernel for gfx1250 (MI455X, wave32, WMMA)
//   scores = (x @ (mu * softplus(sigma))^T) / sqrt(D_IN)
//   lin    =  x @ mu^T
//   out    = relu(scores - gate) * lin + bias
// d_out = [final_output | scores | output], each 8192x8192 fp32.
// 128x128 block tile, 8 wave32s (wave tile 32x64 => 16 WMMA/stage),
// CDNA5 async global->LDS staging (ASYNCcnt), XOR-toggled double buffer,
// non-unrolled stage loop so accumulators keep fixed registers.
// ---------------------------------------------------------------------------

typedef __attribute__((ext_vector_type(16))) __bf16 v16bf;
typedef __attribute__((ext_vector_type(8)))  float  v8f;

constexpr int   DIN   = 2048;
constexpr int   DOUT  = 8192;
constexpr int   NROW  = 4 * 2048;                 // B*S = 8192
constexpr size_t NTOT = (size_t)NROW * DOUT;      // 67,108,864

constexpr int BM = 128, BN = 128, BK = 32;
constexpr int PITCH   = 40;                       // u16 per LDS row: 32 + 8 pad (80B = 20 banks)
constexpr int NSTAGE  = DIN / BK;                 // 64
constexpr int A_OFF   = 0;                        // u16 offsets within one LDS buffer
constexpr int M_OFF   = BM * PITCH;               // 5120
constexpr int K_OFF   = M_OFF + BN * PITCH;       // 10240
constexpr int BUFH    = K_OFF + BN * PITCH;       // 15360 u16 = 30720 B used per buffer
constexpr uint32_t BUFP = 32768;                  // padded buffer stride (bytes, pow2 for XOR)

__device__ __forceinline__ uint16_t f2bf(float f) {
  uint32_t u = __float_as_uint(f);
  u += 0x7FFFu + ((u >> 16) & 1u);                // round-to-nearest-even
  return (uint16_t)(u >> 16);
}
__device__ __forceinline__ uint32_t pack2(float lo, float hi) {
  return (uint32_t)f2bf(lo) | ((uint32_t)f2bf(hi) << 16);
}
__device__ __forceinline__ float softplus_f(float v) {
  return fmaxf(v, 0.0f) + log1pf(__expf(-fabsf(v)));
}

// ----------------------------- prep kernels --------------------------------

__global__ void prep_x_kernel(const float* __restrict__ x, uint16_t* __restrict__ xb) {
  size_t t = (size_t)blockIdx.x * blockDim.x + threadIdx.x;   // 8 elems / thread
  const float4* p = (const float4*)x + 2 * t;
  float4 a = p[0], b = p[1];
  uint4 o;
  o.x = pack2(a.x, a.y); o.y = pack2(a.z, a.w);
  o.z = pack2(b.x, b.y); o.w = pack2(b.z, b.w);
  ((uint4*)xb)[t] = o;
}

__global__ void prep_w_kernel(const float* __restrict__ mu, const float* __restrict__ sg,
                              uint16_t* __restrict__ mub, uint16_t* __restrict__ keb) {
  size_t t = (size_t)blockIdx.x * blockDim.x + threadIdx.x;   // 8 elems / thread
  const float4* pm = (const float4*)mu + 2 * t;
  const float4* ps = (const float4*)sg + 2 * t;
  float4 m0 = pm[0], m1 = pm[1];
  float4 s0 = ps[0], s1 = ps[1];
  uint4 om, ok;
  om.x = pack2(m0.x, m0.y); om.y = pack2(m0.z, m0.w);
  om.z = pack2(m1.x, m1.y); om.w = pack2(m1.z, m1.w);
  ok.x = pack2(m0.x * softplus_f(s0.x), m0.y * softplus_f(s0.y));
  ok.y = pack2(m0.z * softplus_f(s0.z), m0.w * softplus_f(s0.w));
  ok.z = pack2(m1.x * softplus_f(s1.x), m1.y * softplus_f(s1.y));
  ok.w = pack2(m1.z * softplus_f(s1.z), m1.w * softplus_f(s1.w));
  ((uint4*)mub)[t] = om;
  ((uint4*)keb)[t] = ok;
}

// --------------------------- CDNA5 async staging ---------------------------
__device__ __forceinline__ void async_copy_b128(uint32_t lds_addr, const uint16_t* g) {
  asm volatile("global_load_async_to_lds_b128 %0, %1, off"
               :: "v"(lds_addr), "v"((unsigned long long)(uintptr_t)g)
               : "memory");
}
__device__ __forceinline__ void wait_async0() {
  asm volatile("s_wait_asynccnt 0x0" ::: "memory");
}

// --------------------------- fragment builders -----------------------------
// ISA 16-bit A 16x32 layout: lane L<16 -> row M=L, K={0..7}u{16..23}; lane>=16 K={8..15}u{24..31}
//   this code uses contiguous k = 16h+s instead: chunks at (h*16) and (h*16 + 8) u16, gap 8.
// B 32x16 as used here (same contiguous k as A): lane L<16 -> col N=L, K=0..15; lane>=16 K=16..31
//   => per lane: chunks at (h*16) and (h*16 + 8) u16, gap 8.
__device__ __forceinline__ v16bf frag(const uint16_t* p, int gap) {
  union { uint4 q[2]; v16bf v; } u;
  u.q[0] = *(const uint4*)(p);
  u.q[1] = *(const uint4*)(p + gap);
  return u.v;
}

#define WMMA_BF16(A_, B_, C_) \
  __builtin_amdgcn_wmma_f32_16x16x32_bf16(false, (A_), false, (B_), (short)0, (C_), false, false)

// ------------------------------ fused GEMM ---------------------------------

__global__ __launch_bounds__(256)
void gemm_fused_kernel(const uint16_t* __restrict__ xb,
                       const uint16_t* __restrict__ mub,
                       const uint16_t* __restrict__ keb,
                       const float* __restrict__ gate,
                       const float* __restrict__ bias,
                       float* __restrict__ out) {
  // buffer0 at byte 0, buffer1 at byte BUFP (array base is LDS offset 0)
  __shared__ __align__(16) uint16_t lds[BUFP / 2 + BUFH];

  const int tid = threadIdx.x;
  const int bn  = blockIdx.x;           // 0..63  (DOUT/BN)
  const int bm  = blockIdx.y;           // 0..63  (NROW/BM)

  // global staging: per thread, 2 chunks of A (rows t/4, t/4+64) + 2 of each B
  const int aRow = tid >> 2;
  const int cOff = (tid & 3) * 8;                                   // u16 units
  const uint16_t* gA0 = xb  + ((size_t)(bm * BM + aRow)) * DIN + cOff;
  const uint16_t* gA1 = gA0 + (size_t)64 * DIN;
  const uint16_t* gM0 = mub + ((size_t)(bn * BN + aRow)) * DIN + cOff;
  const uint16_t* gM1 = gM0 + (size_t)64 * DIN;
  const uint16_t* gK0 = keb + ((size_t)(bn * BN + aRow)) * DIN + cOff;
  const uint16_t* gK1 = gK0 + (size_t)64 * DIN;

  // per-thread LDS destinations (byte addresses within buffer 0)
  constexpr uint32_t HALF = (uint32_t)(64 * PITCH * 2);
  const uint32_t dA0 = (uint32_t)(uintptr_t)&lds[A_OFF + aRow * PITCH + cOff];
  const uint32_t dA1 = dA0 + HALF;
  const uint32_t dM0 = (uint32_t)(uintptr_t)&lds[M_OFF + aRow * PITCH + cOff];
  const uint32_t dM1 = dM0 + HALF;
  const uint32_t dK0 = (uint32_t)(uintptr_t)&lds[K_OFF + aRow * PITCH + cOff];
  const uint32_t dK1 = dK0 + HALF;

  const int lane = tid & 31;
  const int wm   = (tid >> 5) >> 1;     // 0..3  wave M (rows wm*32..+31)
  const int wn   = (tid >> 5) & 1;      // 0..1  wave N (cols wn*64..+63)
  const int r    = lane & 15;
  const int h    = lane >> 4;

  // per-wave fragment offsets (u16 units within a buffer)
  const int aO0 = A_OFF + (wm * 32 +      r) * PITCH + h * 16;
  const int aO1 = A_OFF + (wm * 32 + 16 + r) * PITCH + h * 16;
  const int bO  = (wn * 64 + r) * PITCH + h * 16;   // + n*16*PITCH per N-tile

  // named accumulators: only ever written by WMMA (tied C/D) until epilogue
  v8f s00 = {}, s01 = {}, s02 = {}, s03 = {};   // scores, A-tile 0
  v8f s10 = {}, s11 = {}, s12 = {}, s13 = {};   // scores, A-tile 1
  v8f l00 = {}, l01 = {}, l02 = {}, l03 = {};   // linear, A-tile 0
  v8f l10 = {}, l11 = {}, l12 = {}, l13 = {};   // linear, A-tile 1

  // prologue: stage 0 -> buffer 0 (async)
  async_copy_b128(dA0, gA0);
  async_copy_b128(dA1, gA1);
  async_copy_b128(dM0, gM0);
  async_copy_b128(dM1, gM1);
  async_copy_b128(dK0, gK0);
  async_copy_b128(dK1, gK1);

  uint32_t rdo = 0;                     // read-buffer byte offset
  uint32_t wro = BUFP;                  // write-buffer byte offset

#pragma clang loop unroll(disable)
  for (int s = 0; s < NSTAGE; ++s) {
    wait_async0();                      // my async writes (stage s) have landed
    __syncthreads();                    // everyone's writes landed / reads done

    if (s + 1 < NSTAGE) {               // stage s+1 -> other buffer (async)
      const size_t k0 = (size_t)(s + 1) * BK;
      async_copy_b128(dA0 + wro, gA0 + k0);
      async_copy_b128(dA1 + wro, gA1 + k0);
      async_copy_b128(dM0 + wro, gM0 + k0);
      async_copy_b128(dM1 + wro, gM1 + k0);
      async_copy_b128(dK0 + wro, gK0 + k0);
      async_copy_b128(dK1 + wro, gK1 + k0);
    }

    const uint16_t* base = (const uint16_t*)((const char*)lds + rdo);
    v16bf a0 = frag(base + aO0, 8);
    v16bf a1 = frag(base + aO1, 8);

    {
      v16bf bK = frag(base + K_OFF + bO, 8);
      v16bf bM = frag(base + M_OFF + bO, 8);
      s00 = WMMA_BF16(a0, bK, s00);  s10 = WMMA_BF16(a1, bK, s10);
      l00 = WMMA_BF16(a0, bM, l00);  l10 = WMMA_BF16(a1, bM, l10);
    }
    {
      v16bf bK = frag(base + K_OFF + bO + 16 * PITCH, 8);
      v16bf bM = frag(base + M_OFF + bO + 16 * PITCH, 8);
      s01 = WMMA_BF16(a0, bK, s01);  s11 = WMMA_BF16(a1, bK, s11);
      l01 = WMMA_BF16(a0, bM, l01);  l11 = WMMA_BF16(a1, bM, l11);
    }
    {
      v16bf bK = frag(base + K_OFF + bO + 32 * PITCH, 8);
      v16bf bM = frag(base + M_OFF + bO + 32 * PITCH, 8);
      s02 = WMMA_BF16(a0, bK, s02);  s12 = WMMA_BF16(a1, bK, s12);
      l02 = WMMA_BF16(a0, bM, l02);  l12 = WMMA_BF16(a1, bM, l12);
    }
    {
      v16bf bK = frag(base + K_OFF + bO + 48 * PITCH, 8);
      v16bf bM = frag(base + M_OFF + bO + 48 * PITCH, 8);
      s03 = WMMA_BF16(a0, bK, s03);  s13 = WMMA_BF16(a1, bK, s13);
      l03 = WMMA_BF16(a0, bM, l03);  l13 = WMMA_BF16(a1, bM, l13);
    }

    rdo ^= BUFP;
    wro ^= BUFP;
  }

  // epilogue: scale, gate, combine; C/D layout: lane -> N=r, VGPR j -> M=j+8*h
  constexpr float SCALE = 0.022097086912079608f;   // 1/sqrt(2048)
  const int colBase = bn * BN + wn * 64 + r;
  const int rowBase = bm * BM + wm * 32 + h * 8;

  auto emit = [&](const v8f& aS, const v8f& aL, int row0, int col) {
    const float g  = gate[col];
    const float bi = bias[col];
#pragma unroll
    for (int j = 0; j < 8; ++j) {
      const float sc = aS[j] * SCALE;
      const float li = aL[j];
      const float rw = fmaxf(sc - g, 0.0f);
      const float o  = fmaf(rw, li, bi);
      const size_t idx = (size_t)(row0 + j) * DOUT + col;
      out[idx]            = o;   // final_output
      out[NTOT + idx]     = sc;  // scores
      out[2 * NTOT + idx] = o;   // output
    }
  };

  emit(s00, l00, rowBase,      colBase);
  emit(s01, l01, rowBase,      colBase + 16);
  emit(s02, l02, rowBase,      colBase + 32);
  emit(s03, l03, rowBase,      colBase + 48);
  emit(s10, l10, rowBase + 16, colBase);
  emit(s11, l11, rowBase + 16, colBase + 16);
  emit(s12, l12, rowBase + 16, colBase + 32);
  emit(s13, l13, rowBase + 16, colBase + 48);
}

// ------------------------------- launcher ----------------------------------

extern "C" void kernel_launch(void* const* d_in, const int* in_sizes, int n_in,
                              void* d_out, int out_size, void* d_ws, size_t ws_size,
                              hipStream_t stream) {
  const float* x    = (const float*)d_in[0];
  const float* mu   = (const float*)d_in[1];
  const float* sg   = (const float*)d_in[2];
  const float* gate = (const float*)d_in[3];
  const float* bias = (const float*)d_in[4];
  float* out = (float*)d_out;

  // workspace: bf16 copies (all fit in the 192MB L2)
  uint16_t* xb  = (uint16_t*)d_ws;                          // 8192*2048*2B = 32MB
  uint16_t* mub = xb  + (size_t)NROW * DIN;                 // 32MB
  uint16_t* keb = mub + (size_t)DOUT * DIN;                 // 32MB

  constexpr int XBLOCKS = (NROW * DIN) / 8 / 256;           // 8192
  constexpr int WBLOCKS = (DOUT * DIN) / 8 / 256;           // 8192
  prep_x_kernel<<<XBLOCKS, 256, 0, stream>>>(x, xb);
  prep_w_kernel<<<WBLOCKS, 256, 0, stream>>>(mu, sg, mub, keb);

  dim3 grid(DOUT / BN, NROW / BM);                          // 64 x 64
  gemm_fused_kernel<<<grid, 256, 0, stream>>>(xb, mub, keb, gate, bias, out);
}
